// EncoderBlock_31817117729042
// MI455X (gfx1250) — compile-verified, matrix operand order checked by reading, not run on hardware
//
#include <hip/hip_runtime.h>
#include <hip/hip_bf16.h>

// ---------------------------------------------------------------------------
// Types for CDNA5 WMMA (wave32, 16x16x32 bf16 -> f32)
// ---------------------------------------------------------------------------
typedef __bf16 bf16_t;
typedef __attribute__((ext_vector_type(16))) __bf16 v16bf;
typedef __attribute__((ext_vector_type(8)))  float  v8f;

union FragBF {
    v16bf v;
    uint4 q[2];   // q[0] -> VGPRs 0-3 (k-chunk at kbase), q[1] -> VGPRs 4-7 (kbase+16)
};

__device__ __forceinline__ v8f wmma_bf16(const FragBF& a, const FragBF& b, v8f c) {
    return __builtin_amdgcn_wmma_f32_16x16x32_bf16(
        /*neg_a=*/false, a.v, /*neg_b=*/false, b.v,
        /*c_mod=*/(short)0, c, /*reuse_a=*/false, /*reuse_b=*/false);
}

// ---------------------------------------------------------------------------
// Tensor Data Mover: 2D tile (32 bf16 wide x `rows` tall) global -> LDS with
// hardware row padding (+8 bf16 per 32-elem row => 40-element LDS rows, the
// bank-conflict-free layout the WMMA fragment reads expect).
// 6-arg builtin: (uint32x4 g0, int32x8 g1, int32x4 g2, int32x4 g3, int32x8 gx, i32 cpol)
// ---------------------------------------------------------------------------
#if __has_builtin(__builtin_amdgcn_tensor_load_to_lds) && __has_builtin(__builtin_amdgcn_s_wait_tensorcnt)
#define USE_TDM 1
#else
#define USE_TDM 0
#endif

#if USE_TDM
typedef __attribute__((ext_vector_type(4))) unsigned int u32x4_t;
typedef __attribute__((ext_vector_type(8))) int          i32x8_t;
typedef __attribute__((ext_vector_type(4))) int          i32x4_t;

__device__ __forceinline__ void tdm_load_tile_2d(uint32_t lds_off, const bf16_t* gptr,
                                                 uint32_t rows, uint32_t tensor_d0,
                                                 uint32_t tensor_d1, uint32_t stride_elems) {
    const uint64_t ga = (uint64_t)(uintptr_t)gptr;
    // --- D# group 0 (ISA 8.3): count=1 | lds_addr | global_addr[56:0] | type=2
    u32x4_t g0;
    g0[0] = 1u;
    g0[1] = lds_off;
    g0[2] = (uint32_t)ga;
    g0[3] = ((uint32_t)(ga >> 32) & 0x01FFFFFFu) | (2u << 30);
    // --- D# group 1 (ISA 8.4):
    // dw0: data_size=1(2B)@16 | pad_enable@20 | pad_interval=3(16 dw)@22 | pad_amount=3(4 dw)@25
    i32x8_t g1;
    g1[0] = (int)((1u << 16) | (1u << 20) | (3u << 22) | (3u << 25));
    g1[1] = (int)((tensor_d0 & 0xFFFFu) << 16);                                   // tensor_dim0[15:0] @bits63:48
    g1[2] = (int)(((tensor_d0 >> 16) & 0xFFFFu) | ((tensor_d1 & 0xFFFFu) << 16)); // dim0[31:16] | dim1[15:0]
    g1[3] = (int)(((tensor_d1 >> 16) & 0xFFFFu) | (32u << 16));                   // dim1[31:16] | tile_dim0=32
    g1[4] = (int)(rows & 0xFFFFu);                                                // tile_dim1 | tile_dim2=0
    g1[5] = (int)stride_elems;                                                    // tensor_dim0_stride[31:0]
    g1[6] = 0;                                                                    // stride[47:32] | dim1_stride lo
    g1[7] = 0;
    i32x4_t z4 = {0, 0, 0, 0};
    i32x8_t z8 = {0, 0, 0, 0, 0, 0, 0, 0};
    __builtin_amdgcn_tensor_load_to_lds(g0, g1, z4, z4, z8, 0);
}
#endif

// ---------------------------------------------------------------------------
// fp32 -> bf16 elementwise cast
// ---------------------------------------------------------------------------
__global__ __launch_bounds__(256) void cast_bf16_kernel(const float* __restrict__ in,
                                                        bf16_t* __restrict__ out, int n) {
    int i = blockIdx.x * 256 + threadIdx.x;
    if (i < n) out[i] = (bf16_t)in[i];
}

// ---------------------------------------------------------------------------
// Pre-norm (torch.std ddof=1, scalar alpha/beta) -> bf16 output, one row/block
// ---------------------------------------------------------------------------
__global__ __launch_bounds__(256) void norm_kernel(const float* __restrict__ x,
                                                   const float* __restrict__ alpha,
                                                   const float* __restrict__ beta,
                                                   bf16_t* __restrict__ out) {
    __shared__ float sh[512];
    const int row = blockIdx.x, tid = threadIdx.x;
    const float* xr = x + (size_t)row * 768;
    float s = 0.f, s2 = 0.f;
    for (int i = tid; i < 768; i += 256) { float v = xr[i]; s += v; s2 += v * v; }
    sh[tid] = s; sh[256 + tid] = s2;
    __syncthreads();
    for (int o = 128; o > 0; o >>= 1) {
        if (tid < o) { sh[tid] += sh[tid + o]; sh[256 + tid] += sh[256 + tid + o]; }
        __syncthreads();
    }
    const float mean = sh[0] * (1.0f / 768.0f);
    const float var  = fmaxf((sh[256] - 768.0f * mean * mean) * (1.0f / 767.0f), 0.0f);
    const float inv  = alpha[0] / (sqrtf(var) + 1e-6f);
    const float bb   = beta[0];
    for (int i = tid; i < 768; i += 256)
        out[(size_t)row * 768 + i] = (bf16_t)((xr[i] - mean) * inv + bb);
}

// ---------------------------------------------------------------------------
// WMMA GEMM:  out[M,N] = A[M,K](bf16) @ W[N,K](bf16)^T + bias (+res) (relu?)
// Block: 256 thr = 8 waves (4x2), block tile 64x128, wave tile 16x64, K-step 32.
// LDS rows are 40 bf16 (80B = 20 banks) so 16-lane b128 fragment reads are
// conflict-free. Tile staging by the Tensor Data Mover, double-buffered:
// wave 0 issues next K-step's D# loads into the idle buffer before the block
// computes the current one; TENSORcnt wait + end-of-step barrier publish it,
// hiding DMA latency under the WMMAs.
// ---------------------------------------------------------------------------
template <bool RELU, bool OUT_BF16, bool HAS_RES>
__global__ __launch_bounds__(256) void gemm_wmma(const bf16_t* __restrict__ A,
                                                 const bf16_t* __restrict__ W,
                                                 const float*  __restrict__ bias,
                                                 const float*  __restrict__ res,
                                                 float*        __restrict__ outF,
                                                 bf16_t*       __restrict__ outH,
                                                 int M, int N, int K) {
    __shared__ __align__(16) bf16_t As[2][64 * 40];
    __shared__ __align__(16) bf16_t Ws[2][128 * 40];
    const int tid  = threadIdx.x;
    const int lane = tid & 31, wave = tid >> 5;
    const int wm = wave >> 1, wn = wave & 1;
    const int l15 = lane & 15, kb = (lane >> 4) * 8;
    const int bm = blockIdx.x * 64, bn = blockIdx.y * 128;

    v8f acc[4] = {};
#if USE_TDM
    const uint32_t ldsA[2] = {(uint32_t)(uintptr_t)&As[0][0], (uint32_t)(uintptr_t)&As[1][0]};
    const uint32_t ldsW[2] = {(uint32_t)(uintptr_t)&Ws[0][0], (uint32_t)(uintptr_t)&Ws[1][0]};
    // prologue: prime buffer 0 with the first K-step tiles
    if (wave == 0) {
        tdm_load_tile_2d(ldsA[0], A + (size_t)bm * K, 64u,
                         (uint32_t)K, (uint32_t)(M - bm), (uint32_t)K);
        tdm_load_tile_2d(ldsW[0], W + (size_t)bn * K, 128u,
                         (uint32_t)K, (uint32_t)(N - bn), (uint32_t)K);
        __builtin_amdgcn_s_wait_tensorcnt(0);
    }
    __syncthreads();
    int cur = 0;
    for (int k0 = 0; k0 < K; k0 += 32) {
        if (wave == 0 && k0 + 32 < K) {  // async: stage next tiles into idle buffer
            tdm_load_tile_2d(ldsA[cur ^ 1], A + (size_t)bm * K + k0 + 32, 64u,
                             (uint32_t)(K - k0 - 32), (uint32_t)(M - bm), (uint32_t)K);
            tdm_load_tile_2d(ldsW[cur ^ 1], W + (size_t)bn * K + k0 + 32, 128u,
                             (uint32_t)(K - k0 - 32), (uint32_t)(N - bn), (uint32_t)K);
        }
        const bf16_t* Asb = &As[cur][0];
        const bf16_t* Wsb = &Ws[cur][0];
        FragBF a;
        a.q[0] = *(const uint4*)&Asb[(wm * 16 + l15) * 40 + kb];
        a.q[1] = *(const uint4*)&Asb[(wm * 16 + l15) * 40 + kb + 16];
#pragma unroll
        for (int nt = 0; nt < 4; ++nt) {
            FragBF bfr;
            const int rw = wn * 64 + nt * 16 + l15;
            bfr.q[0] = *(const uint4*)&Wsb[rw * 40 + kb];
            bfr.q[1] = *(const uint4*)&Wsb[rw * 40 + kb + 16];
            acc[nt] = wmma_bf16(a, bfr, acc[nt]);
        }
        if (wave == 0) __builtin_amdgcn_s_wait_tensorcnt(0);  // next tiles landed
        __syncthreads();  // publish next buffer + certify current buffer free
        cur ^= 1;
    }
#else
    const int ar = tid >> 2, ac = (tid & 3) * 8;
    for (int k0 = 0; k0 < K; k0 += 32) {
        __syncthreads();
        *(uint4*)&As[0][ar * 40 + ac]        = *(const uint4*)&A[(size_t)(bm + ar) * K + k0 + ac];
        *(uint4*)&Ws[0][ar * 40 + ac]        = *(const uint4*)&W[(size_t)(bn + ar) * K + k0 + ac];
        *(uint4*)&Ws[0][(ar + 64) * 40 + ac] = *(const uint4*)&W[(size_t)(bn + ar + 64) * K + k0 + ac];
        if (k0 + 32 < K) {
            __builtin_prefetch(&A[(size_t)(bm + ar) * K + k0 + 32 + ac], 0, 1);
            __builtin_prefetch(&W[(size_t)(bn + ar) * K + k0 + 32 + ac], 0, 1);
        }
        __syncthreads();
        FragBF a;
        a.q[0] = *(const uint4*)&As[0][(wm * 16 + l15) * 40 + kb];
        a.q[1] = *(const uint4*)&As[0][(wm * 16 + l15) * 40 + kb + 16];
#pragma unroll
        for (int nt = 0; nt < 4; ++nt) {
            FragBF bfr;
            const int rw = wn * 64 + nt * 16 + l15;
            bfr.q[0] = *(const uint4*)&Ws[0][rw * 40 + kb];
            bfr.q[1] = *(const uint4*)&Ws[0][rw * 40 + kb + 16];
            acc[nt] = wmma_bf16(a, bfr, acc[nt]);
        }
    }
#endif
    const int half8 = (lane >> 4) * 8;
#pragma unroll
    for (int nt = 0; nt < 4; ++nt) {
        const int col = bn + wn * 64 + nt * 16 + l15;
        const float bv = bias[col];
#pragma unroll
        for (int r = 0; r < 8; ++r) {
            const int row = bm + wm * 16 + half8 + r;
            float v = acc[nt][r] + bv;
            if (RELU)    v = fmaxf(v, 0.0f);
            if (HAS_RES) v += res[(size_t)row * N + col];
            if (OUT_BF16) outH[(size_t)row * N + col] = (bf16_t)v;
            else          outF[(size_t)row * N + col] = v;
        }
    }
}

// ---------------------------------------------------------------------------
// Flash-style attention with WMMA. Grid: (S/128, B*H). Block: 8 waves, each
// wave owns 16 q-rows; k-blocks of 64 keys. Q/K fragments load directly from
// global (contiguous rows); V is transposed into LDS so the PV B-fragments
// are contiguous ds_load_b128. Online softmax via width-16 shuffles (a C-tile
// row lives in one 16-lane half).
// ---------------------------------------------------------------------------
__global__ __launch_bounds__(256) void attn_wmma(const bf16_t* __restrict__ q,
                                                 const bf16_t* __restrict__ k,
                                                 const bf16_t* __restrict__ v,
                                                 const int*    __restrict__ mask,
                                                 bf16_t*       __restrict__ ctx) {
    const int S = 4096, D = 768, H = 12;
    const int bh = blockIdx.y;
    const int b = bh / H, h = bh % H;
    const int q0 = blockIdx.x * 128;
    const int tid = threadIdx.x, lane = tid & 31, wave = tid >> 5;
    const int l15 = lane & 15, kb = (lane >> 4) * 8, half8 = (lane >> 4) * 8;
    __shared__ __align__(16) bf16_t Vt[64 * 72];        // V^T tile [d][kk]
    __shared__ __align__(16) bf16_t Pst[8 * 16 * 72];   // per-wave P tile [16][64]

    const size_t rowbase = (size_t)b * S;
    FragBF qa[2];
    {
        const bf16_t* qp = q + (rowbase + q0 + wave * 16 + l15) * D + h * 64;
        qa[0].q[0] = *(const uint4*)(qp + kb);
        qa[0].q[1] = *(const uint4*)(qp + kb + 16);
        qa[1].q[0] = *(const uint4*)(qp + 32 + kb);
        qa[1].q[1] = *(const uint4*)(qp + 32 + kb + 16);
    }
    v8f o[4] = {};
    float mrun[8], lrun[8];
#pragma unroll
    for (int r = 0; r < 8; ++r) { mrun[r] = -3.0e38f; lrun[r] = 0.0f; }
    const float scale = 0.125f;  // 1/sqrt(64)
    const int* mrow = mask + b * S;
    bf16_t* P = &Pst[wave * 16 * 72];

    for (int kblk = 0; kblk < S; kblk += 64) {
        __syncthreads();  // protect Vt readers of previous iteration
#pragma unroll
        for (int i = 0; i < 16; ++i) {          // cooperative V transpose
            const int idx = tid + 256 * i;
            const int kk = idx >> 6, d = idx & 63;
            Vt[d * 72 + kk] = v[(rowbase + kblk + kk) * D + h * 64 + d];
        }
        __syncthreads();
        // --- S = Q K^T (4 tiles of 16x16 per wave) ---
        v8f s4[4];
#pragma unroll
        for (int nt = 0; nt < 4; ++nt) {
            const bf16_t* kp = k + (rowbase + kblk + nt * 16 + l15) * D + h * 64;
            FragBF b0, b1;
            b0.q[0] = *(const uint4*)(kp + kb);      b0.q[1] = *(const uint4*)(kp + kb + 16);
            b1.q[0] = *(const uint4*)(kp + 32 + kb); b1.q[1] = *(const uint4*)(kp + 32 + kb + 16);
            v8f t = {};
            t = wmma_bf16(qa[0], b0, t);
            t = wmma_bf16(qa[1], b1, t);
            s4[nt] = t;
        }
        // --- mask + scale, running row max ---
        float mnew[8];
#pragma unroll
        for (int r = 0; r < 8; ++r) mnew[r] = mrun[r];
#pragma unroll
        for (int nt = 0; nt < 4; ++nt) {
            const int mval = mrow[kblk + nt * 16 + l15];
#pragma unroll
            for (int r = 0; r < 8; ++r) {
                float sv = s4[nt][r] * scale;
                if (mval == 0) sv = -1.0e9f;
                s4[nt][r] = sv;
                mnew[r] = fmaxf(mnew[r], sv);
            }
        }
#pragma unroll
        for (int off = 8; off > 0; off >>= 1)
#pragma unroll
            for (int r = 0; r < 8; ++r)
                mnew[r] = fmaxf(mnew[r], __shfl_xor(mnew[r], off, 16));
        float alpha_[8], rs[8];
#pragma unroll
        for (int r = 0; r < 8; ++r) {
            alpha_[r] = __expf(mrun[r] - mnew[r]);
            mrun[r] = mnew[r];
            rs[r] = 0.0f;
        }
#pragma unroll
        for (int nt = 0; nt < 4; ++nt)
#pragma unroll
            for (int r = 0; r < 8; ++r) {
                const float p = __expf(s4[nt][r] - mrun[r]);
                s4[nt][r] = p;
                rs[r] += p;
            }
#pragma unroll
        for (int off = 8; off > 0; off >>= 1)
#pragma unroll
            for (int r = 0; r < 8; ++r)
                rs[r] += __shfl_xor(rs[r], off, 16);
#pragma unroll
        for (int r = 0; r < 8; ++r) lrun[r] = lrun[r] * alpha_[r] + rs[r];
#pragma unroll
        for (int nt = 0; nt < 4; ++nt)
#pragma unroll
            for (int r = 0; r < 8; ++r) o[nt][r] *= alpha_[r];
        // --- relayout P (C-layout) -> A-layout via per-wave LDS tile ---
#pragma unroll
        for (int nt = 0; nt < 4; ++nt)
#pragma unroll
            for (int r = 0; r < 8; ++r)
                P[(half8 + r) * 72 + nt * 16 + l15] = (bf16_t)s4[nt][r];
        __syncthreads();
        FragBF pa0, pa1;
        pa0.q[0] = *(const uint4*)&P[l15 * 72 + kb];
        pa0.q[1] = *(const uint4*)&P[l15 * 72 + kb + 16];
        pa1.q[0] = *(const uint4*)&P[l15 * 72 + 32 + kb];
        pa1.q[1] = *(const uint4*)&P[l15 * 72 + 32 + kb + 16];
        // --- O += P V  (B-fragments from transposed Vt) ---
#pragma unroll
        for (int dt = 0; dt < 4; ++dt) {
            const bf16_t* vp = &Vt[(dt * 16 + l15) * 72];
            FragBF b0, b1;
            b0.q[0] = *(const uint4*)(vp + kb);      b0.q[1] = *(const uint4*)(vp + kb + 16);
            b1.q[0] = *(const uint4*)(vp + 32 + kb); b1.q[1] = *(const uint4*)(vp + 32 + kb + 16);
            o[dt] = wmma_bf16(pa0, b0, o[dt]);
            o[dt] = wmma_bf16(pa1, b1, o[dt]);
        }
    }
    // --- normalize by l, store ctx as bf16 in [B,S,H*dk] layout ---
#pragma unroll
    for (int dt = 0; dt < 4; ++dt)
#pragma unroll
        for (int r = 0; r < 8; ++r) {
            const float val = o[dt][r] / fmaxf(lrun[r], 1.0e-20f);
            const int row = q0 + wave * 16 + half8 + r;
            ctx[(rowbase + row) * D + h * 64 + dt * 16 + l15] = (bf16_t)val;
        }
}

// ---------------------------------------------------------------------------
// Host orchestration
// ---------------------------------------------------------------------------
extern "C" void kernel_launch(void* const* d_in, const int* in_sizes, int n_in,
                              void* d_out, int out_size, void* d_ws, size_t ws_size,
                              hipStream_t stream) {
    (void)in_sizes; (void)n_in; (void)out_size; (void)ws_size;
    const int B = 2, S = 4096, D = 768, DFF = 3072;
    const int M = B * S;

    const float* x    = (const float*)d_in[0];
    const int*   mask = (const int*)  d_in[1];
    const float* wq   = (const float*)d_in[2];
    const float* bq   = (const float*)d_in[3];
    const float* wk   = (const float*)d_in[4];
    const float* bk   = (const float*)d_in[5];
    const float* wv   = (const float*)d_in[6];
    const float* bv   = (const float*)d_in[7];
    const float* wo   = (const float*)d_in[8];
    const float* bo   = (const float*)d_in[9];
    const float* w1   = (const float*)d_in[10];
    const float* b1   = (const float*)d_in[11];
    const float* w2   = (const float*)d_in[12];
    const float* b2   = (const float*)d_in[13];
    const float* alpha1 = (const float*)d_in[14];
    const float* beta1  = (const float*)d_in[15];
    const float* alpha2 = (const float*)d_in[16];
    const float* beta2  = (const float*)d_in[17];
    float* out = (float*)d_out;

    char* ws = (char*)d_ws;
    size_t off = 0;
    auto alloc = [&](size_t bytes) -> char* {
        char* p = ws + off;
        off += (bytes + 255) & ~(size_t)255;
        return p;
    };
    bf16_t* wqh = (bf16_t*)alloc((size_t)D * D * 2);
    bf16_t* wkh = (bf16_t*)alloc((size_t)D * D * 2);
    bf16_t* wvh = (bf16_t*)alloc((size_t)D * D * 2);
    bf16_t* woh = (bf16_t*)alloc((size_t)D * D * 2);
    bf16_t* w1h = (bf16_t*)alloc((size_t)DFF * D * 2);
    bf16_t* w2h = (bf16_t*)alloc((size_t)D * DFF * 2);
    bf16_t* h1  = (bf16_t*)alloc((size_t)M * D * 2);   // reused as ctx
    bf16_t* qb  = (bf16_t*)alloc((size_t)M * D * 2);   // reused as h2
    bf16_t* kbf = (bf16_t*)alloc((size_t)M * D * 2);
    bf16_t* vbf = (bf16_t*)alloc((size_t)M * D * 2);
    float*  x1  = (float*) alloc((size_t)M * D * 4);
    bf16_t* ff1 = (bf16_t*)alloc((size_t)M * DFF * 2);
    bf16_t* ctx = h1;   // h1 dead after QKV projections
    bf16_t* h2  = qb;   // q dead after attention

    // weight casts fp32 -> bf16 (layout [N,K] row-major == WMMA B fragments)
    cast_bf16_kernel<<<(D * D + 255) / 256, 256, 0, stream>>>(wq, wqh, D * D);
    cast_bf16_kernel<<<(D * D + 255) / 256, 256, 0, stream>>>(wk, wkh, D * D);
    cast_bf16_kernel<<<(D * D + 255) / 256, 256, 0, stream>>>(wv, wvh, D * D);
    cast_bf16_kernel<<<(D * D + 255) / 256, 256, 0, stream>>>(wo, woh, D * D);
    cast_bf16_kernel<<<(DFF * D + 255) / 256, 256, 0, stream>>>(w1, w1h, DFF * D);
    cast_bf16_kernel<<<(D * DFF + 255) / 256, 256, 0, stream>>>(w2, w2h, D * DFF);

    // residual 0: pre-norm -> QKV -> attention -> O-proj + residual
    norm_kernel<<<M, 256, 0, stream>>>(x, alpha1, beta1, h1);
    dim3 gD(M / 64, D / 128);
    gemm_wmma<false, true, false><<<gD, 256, 0, stream>>>(h1, wqh, bq, nullptr, nullptr, qb,  M, D, D);
    gemm_wmma<false, true, false><<<gD, 256, 0, stream>>>(h1, wkh, bk, nullptr, nullptr, kbf, M, D, D);
    gemm_wmma<false, true, false><<<gD, 256, 0, stream>>>(h1, wvh, bv, nullptr, nullptr, vbf, M, D, D);
    attn_wmma<<<dim3(S / 128, B * 12), 256, 0, stream>>>(qb, kbf, vbf, mask, ctx);
    gemm_wmma<false, false, true><<<gD, 256, 0, stream>>>(ctx, woh, bo, x, x1, nullptr, M, D, D);

    // residual 1: pre-norm -> FFN (relu) + residual
    norm_kernel<<<M, 256, 0, stream>>>(x1, alpha2, beta2, h2);
    gemm_wmma<true, true, false><<<dim3(M / 64, DFF / 128), 256, 0, stream>>>(h2, w1h, b1, nullptr, nullptr, ff1, M, DFF, D);
    gemm_wmma<false, false, true><<<gD, 256, 0, stream>>>(ff1, w2h, b2, x1, out, nullptr, M, D, DFF);
}
